// SSMBlock_86766929313980
// MI455X (gfx1250) — matrix operand order repaired and checked by reading, not compiled, not run on hardware
//
#include <hip/hip_runtime.h>
#include <hip/hip_bf16.h>
#include <math.h>
#include <stdint.h>

// ---------------------------------------------------------------------------
// SSM block for MI455X (gfx1250, wave32, WMMA).
// GEMMs: v_wmma_f32_16x16x32_bf16, double-buffered LDS, A-tile staged with
// global_load_async_to_lds_b128 (ASYNCcnt), B-tile register-staged transpose.
// Selective scan: 3-pass chunked parallel scan (dA is scalar per step).
// ---------------------------------------------------------------------------

typedef __attribute__((ext_vector_type(16))) __bf16 v16bf;
typedef __attribute__((ext_vector_type(8)))  float  v8f;

#define BB 2
#define NN 4096
#define DD 1024
#define DSD 1024
#define HH 4096
#define MM (BB * NN)        // 8192 rows for all GEMMs
#define CHUNK 64
#define NCH (NN / CHUNK)    // 64 chunks

#define BM 128
#define BN 128
#define BK 32
#define LDSTRIDE 40         // padded LDS row stride (elements) -> conflict-free b128

static __device__ __forceinline__ float siluf(float v) {
    return v / (1.0f + __expf(-v));
}
static __device__ __forceinline__ float geluf(float v) {
    return 0.5f * v * (1.0f + erff(v * 0.70710678118654752f));
}

enum { EPI_F32 = 0, EPI_GATE = 1, EPI_GELU_BF16 = 2 };

// ---------------------------------------------------------------------------
// bf16 WMMA GEMM: C(MxN) = A(MxK,bf16) @ W(KxN,bf16) + bias (+epilogue)
// 256 threads = 8 wave32; block tile 128x128x32; wave tile 32x64.
// ---------------------------------------------------------------------------
template <int EPI>
__global__ __launch_bounds__(256)
void gemm_bf16_k(const __bf16* __restrict__ A, const __bf16* __restrict__ W,
                 const float* __restrict__ bias, const float* __restrict__ resid,
                 const __bf16* __restrict__ xn,
                 void* __restrict__ out0, void* __restrict__ out1,
                 int M, int N, int K) {
    __shared__ __bf16 As[2][BM][LDSTRIDE];   // row-major A tiles (double buffer)
    __shared__ __bf16 Bs[2][BN][LDSTRIDE];   // B tiles transposed: [n][k]

    const int t    = threadIdx.x;
    const int lane = t & 31;
    const int wid  = t >> 5;
    const int wm   = wid & 3;    // 4 waves along M (32 rows each)
    const int wn   = wid >> 2;   // 2 waves along N (64 cols each)
    const int m0   = blockIdx.x * BM;
    const int n0   = blockIdx.y * BN;

    // global<->LDS staging maps (each thread: 2x16B for A, 2x16B for B)
    const int ar0 = t >> 2;             // A rows ar0, ar0+64
    const int ak0 = (t & 3) << 3;       // A k-offset 0,8,16,24
    const int kr0 = t >> 4;             // B k-rows kr0, kr0+16
    const int nq0 = (t & 15) << 3;      // B col group 0..120

    v8f zero = {0.f, 0.f, 0.f, 0.f, 0.f, 0.f, 0.f, 0.f};
    v8f acc[2][4];
#pragma unroll
    for (int mf = 0; mf < 2; ++mf)
#pragma unroll
        for (int nf = 0; nf < 4; ++nf) acc[mf][nf] = zero;

    // --- async copy of A tile straight into LDS (no VGPR round trip) ---
    auto issueA = [&](int kb, int bufI) {
        uint32_t l0 = (uint32_t)(uintptr_t)&As[bufI][ar0][ak0];
        uint32_t l1 = (uint32_t)(uintptr_t)&As[bufI][ar0 + 64][ak0];
        const __bf16* g0 = A + (size_t)(m0 + ar0) * K + kb + ak0;
        const __bf16* g1 = A + (size_t)(m0 + ar0 + 64) * K + kb + ak0;
        asm volatile("global_load_async_to_lds_b128 %0, %2, off\n\t"
                     "global_load_async_to_lds_b128 %1, %3, off"
                     :: "v"(l0), "v"(l1), "v"(g0), "v"(g1)
                     : "memory");
    };
    auto waitAsync = [&]() {
        asm volatile("s_wait_asynccnt 0x0" ::: "memory");
    };
    // --- B tile: register-staged (needs transpose into Bs[n][k]) ---
    auto loadB = [&](int kb, uint4& v0, uint4& v1) {
        v0 = *(const uint4*)(W + (size_t)(kb + kr0) * N + n0 + nq0);
        v1 = *(const uint4*)(W + (size_t)(kb + kr0 + 16) * N + n0 + nq0);
    };
    auto storeB = [&](int bufI, uint4 v0, uint4 v1) {
        const __bf16* e0 = (const __bf16*)&v0;
        const __bf16* e1 = (const __bf16*)&v1;
#pragma unroll
        for (int j = 0; j < 8; ++j) {
            Bs[bufI][nq0 + j][kr0]      = e0[j];
            Bs[bufI][nq0 + j][kr0 + 16] = e1[j];
        }
    };
    // --- fragment loads, contiguous k order (k = 16*(lane/16) + s) for A and B + 8 WMMAs ---
    auto compute = [&](int bufI) {
        union { v16bf v; uint4 q[2]; } af[2], bf[4];
        const int mrow = lane & 15;
        const int kha  = (lane >> 4) << 4;   // A: lanes<16 K 0..15; else 16..31
#pragma unroll
        for (int mf = 0; mf < 2; ++mf) {
            int row = wm * 32 + mf * 16 + mrow;
            af[mf].q[0] = *(const uint4*)&As[bufI][row][kha];
            af[mf].q[1] = *(const uint4*)&As[bufI][row][kha + 8];
        }
        const int khb = (lane >> 4) << 4;    // B: lanes<16 K 0..15; else 16..31
#pragma unroll
        for (int nf = 0; nf < 4; ++nf) {
            int col = wn * 64 + nf * 16 + mrow;
            bf[nf].q[0] = *(const uint4*)&Bs[bufI][col][khb];
            bf[nf].q[1] = *(const uint4*)&Bs[bufI][col][khb + 8];
        }
#pragma unroll
        for (int mf = 0; mf < 2; ++mf)
#pragma unroll
            for (int nf = 0; nf < 4; ++nf)
                acc[mf][nf] = __builtin_amdgcn_wmma_f32_16x16x32_bf16(
                    false, af[mf].v, false, bf[nf].v, (short)0, acc[mf][nf],
                    false, false);
    };

    // --- software-pipelined main loop (double buffer) ---
    uint4 bv0, bv1;
    issueA(0, 0);
    loadB(0, bv0, bv1);
    storeB(0, bv0, bv1);
    waitAsync();
    __syncthreads();

    int buf = 0;
    for (int kb = 0; kb < K; kb += BK) {
        const bool more = (kb + BK < K);
        if (more) {
            issueA(kb + BK, buf ^ 1);       // async: global -> LDS, overlaps WMMA
            loadB(kb + BK, bv0, bv1);       // issues now; waited after compute
            if (kb + 2 * BK < K)
                __builtin_prefetch(W + (size_t)(kb + 2 * BK + kr0) * N + n0 + nq0, 0, 3);
        }
        compute(buf);
        if (more) {
            storeB(buf ^ 1, bv0, bv1);
            waitAsync();
            __syncthreads();
        }
        buf ^= 1;
    }

    // --- epilogue: C/D layout, VGPR r -> (M = r + (lane/16)*8, N = lane%16) ---
#pragma unroll
    for (int mf = 0; mf < 2; ++mf)
#pragma unroll
        for (int nf = 0; nf < 4; ++nf) {
            int colL = n0 + wn * 64 + nf * 16 + (lane & 15);
#pragma unroll
            for (int r = 0; r < 8; ++r) {
                int rowL = m0 + wm * 32 + mf * 16 + ((lane >> 4) << 3) + r;
                float v = acc[mf][nf][r] + bias[colL];
                if (EPI == EPI_F32) {
                    float o = v;
                    if (resid) o += resid[(size_t)rowL * N + colL];
                    ((float*)out0)[(size_t)rowL * N + colL] = o;
                } else if (EPI == EPI_GATE) {
                    int half = N >> 1;
                    float s = siluf(v);
                    if (colL < half) {  // xg = xn * silu(gate)
                        float xv = (float)xn[(size_t)rowL * half + colL];
                        ((__bf16*)out0)[(size_t)rowL * half + colL] = (__bf16)(xv * s);
                    } else {            // gg = silu(gate_gate)
                        ((__bf16*)out1)[(size_t)rowL * half + (colL - half)] = (__bf16)s;
                    }
                } else {  // EPI_GELU_BF16
                    ((__bf16*)out0)[(size_t)rowL * N + colL] = (__bf16)geluf(v);
                }
            }
        }
}

// ---------------------------------------------------------------------------
// f32 -> bf16 conversion, 4 elements/thread
// ---------------------------------------------------------------------------
__global__ void cvt_bf16_k(const float* __restrict__ s, __bf16* __restrict__ d, int n) {
    int i = (blockIdx.x * 256 + threadIdx.x) * 4;
    if (i < n) {
        float4 v = *(const float4*)(s + i);
        __bf16 o[4] = {(__bf16)v.x, (__bf16)v.y, (__bf16)v.z, (__bf16)v.w};
        *(uint2*)(d + i) = *(const uint2*)o;
    }
}

// ---------------------------------------------------------------------------
// Row LayerNorm (D=1024), writes bf16. One block (8 wave32) per row.
// ---------------------------------------------------------------------------
__global__ __launch_bounds__(256)
void layernorm_k(const float* __restrict__ x, const float* __restrict__ w,
                 const float* __restrict__ b, __bf16* __restrict__ out) {
    __shared__ float s0[8], s1[8], sm[2];
    const int row = blockIdx.x;
    const float* xr = x + (size_t)row * DD;
    float sum = 0.f, sq = 0.f;
    float vals[4];
#pragma unroll
    for (int j = 0; j < 4; ++j) {
        float v = xr[threadIdx.x + j * 256];
        vals[j] = v; sum += v; sq += v * v;
    }
    for (int off = 16; off; off >>= 1) {
        sum += __shfl_xor(sum, off, 32);
        sq  += __shfl_xor(sq,  off, 32);
    }
    int lane = threadIdx.x & 31, wid = threadIdx.x >> 5;
    if (lane == 0) { s0[wid] = sum; s1[wid] = sq; }
    __syncthreads();
    if (threadIdx.x == 0) {
        float a = 0.f, c = 0.f;
        for (int i = 0; i < 8; ++i) { a += s0[i]; c += s1[i]; }
        float mu  = a / DD;
        float var = c / DD - mu * mu;
        sm[0] = mu; sm[1] = rsqrtf(var + 1e-5f);
    }
    __syncthreads();
    float mu = sm[0], rstd = sm[1];
#pragma unroll
    for (int j = 0; j < 4; ++j) {
        int i = threadIdx.x + j * 256;
        out[(size_t)row * DD + i] = (__bf16)((vals[j] - mu) * rstd * w[i] + b[i]);
    }
}

// ---------------------------------------------------------------------------
// wmean[k] = mean_j state_w[k][j]; wmean[DD] = mean(state_b). grid = DD+1.
// ---------------------------------------------------------------------------
__global__ __launch_bounds__(256)
void wmean_k(const float* __restrict__ W, const float* __restrict__ bias,
             float* __restrict__ wmean) {
    __shared__ float s0[8];
    int k = blockIdx.x;
    float sum = 0.f;
    if (k < DD) {
        for (int j = threadIdx.x; j < DSD; j += 256) sum += W[(size_t)k * DSD + j];
    } else {
        for (int j = threadIdx.x; j < DSD; j += 256) sum += bias[j];
    }
    for (int off = 16; off; off >>= 1) sum += __shfl_xor(sum, off, 32);
    int lane = threadIdx.x & 31, wid = threadIdx.x >> 5;
    if (lane == 0) s0[wid] = sum;
    __syncthreads();
    if (threadIdx.x == 0) {
        float a = 0.f;
        for (int i = 0; i < 8; ++i) a += s0[i];
        wmean[k] = a / (float)DSD;
    }
}

// ---------------------------------------------------------------------------
// delta[row] = softplus(xn[row].wmean + bmean); dA[row] = exp(-delta).
// (mean over DS of xn@state_w+b collapses to a matvec.)
// ---------------------------------------------------------------------------
__global__ __launch_bounds__(256)
void delta_k(const __bf16* __restrict__ xn, const float* __restrict__ wmean,
             float* __restrict__ delta, float* __restrict__ dA) {
    __shared__ float s0[8];
    int row = blockIdx.x;
    float sum = 0.f;
    for (int j = threadIdx.x; j < DD; j += 256)
        sum += (float)xn[(size_t)row * DD + j] * wmean[j];
    for (int off = 16; off; off >>= 1) sum += __shfl_xor(sum, off, 32);
    int lane = threadIdx.x & 31, wid = threadIdx.x >> 5;
    if (lane == 0) s0[wid] = sum;
    __syncthreads();
    if (threadIdx.x == 0) {
        float v = wmean[DD];
        for (int i = 0; i < 8; ++i) v += s0[i];
        float d = (v > 20.f) ? v : log1pf(__expf(v));
        delta[row] = d;
        dA[row]    = __expf(-d);
    }
}

// dBu = delta * Bm * u  (in place over u), 4 floats/thread
__global__ void dbu_k(float* __restrict__ u, const float* __restrict__ Bm,
                      const float* __restrict__ delta) {
    size_t i = ((size_t)blockIdx.x * 256 + threadIdx.x) * 4;
    int row  = (int)(i / DSD);
    float4 uu = *(const float4*)(u + i);
    float4 bb = *(const float4*)(Bm + i);
    float d = delta[row];
    uu.x = d * bb.x * uu.x; uu.y = d * bb.y * uu.y;
    uu.z = d * bb.z * uu.z; uu.w = d * bb.w * uu.w;
    *(float4*)(u + i) = uu;
}

// cumA: per-chunk inclusive cumprod of dA; P = full chunk product. 1 block.
__global__ void cuma_k(const float* __restrict__ dA, float* __restrict__ cumA,
                       float* __restrict__ P) {
    int id = threadIdx.x;            // BB*NCH = 128 threads
    if (id >= BB * NCH) return;
    int b = id / NCH, c = id % NCH;
    int base = b * NN + c * CHUNK;
    float p = 1.f;
    for (int i = 0; i < CHUNK; ++i) { p *= dA[base + i]; cumA[base + i] = p; }
    P[id] = p;
}

// pass 1: local scan within chunk (4 channels/thread); record chunk carry.
__global__ __launch_bounds__(256)
void scan1_k(const float* __restrict__ dBu, const float* __restrict__ dA,
             float* __restrict__ yloc, float* __restrict__ carry) {
    int c  = blockIdx.x;                 // chunk
    int b  = blockIdx.y;                 // batch
    int ch = threadIdx.x * 4;            // 256 threads x 4 = DSD channels
    int base = b * NN + c * CHUNK;
    float4 h = {0.f, 0.f, 0.f, 0.f};
    for (int i = 0; i < CHUNK; ++i) {
        size_t idx = (size_t)(base + i) * DSD + ch;
        float a = dA[base + i];
        float4 d = *(const float4*)(dBu + idx);
        h.x = a * h.x + d.x; h.y = a * h.y + d.y;
        h.z = a * h.z + d.z; h.w = a * h.w + d.w;
        *(float4*)(yloc + idx) = h;
    }
    *(float4*)(carry + (size_t)(b * NCH + c) * DSD + ch) = h;
}

// pass 2: sequential carry across chunks -> H entering each chunk.
__global__ __launch_bounds__(256)
void scan2_k(const float* __restrict__ carry, const float* __restrict__ P,
             float* __restrict__ Hin) {
    int b  = blockIdx.x;
    int ch = threadIdx.x * 4;
    float4 H = {0.f, 0.f, 0.f, 0.f};
    for (int c = 0; c < NCH; ++c) {
        size_t idx = (size_t)(b * NCH + c) * DSD + ch;
        *(float4*)(Hin + idx) = H;
        float p = P[b * NCH + c];
        float4 cv = *(const float4*)(carry + idx);
        H.x = p * H.x + cv.x; H.y = p * H.y + cv.y;
        H.z = p * H.z + cv.z; H.w = p * H.w + cv.w;
    }
}

// pass 3: y = yloc + cumA * Hin, emit bf16 for the out-proj WMMA GEMM.
__global__ void scan3_k(const float* __restrict__ yloc, const float* __restrict__ cumA,
                        const float* __restrict__ Hin, __bf16* __restrict__ ybf) {
    size_t i = ((size_t)blockIdx.x * 256 + threadIdx.x) * 4;
    int row = (int)(i / DSD);
    int ch  = (int)(i % DSD);
    int b = row / NN, n = row % NN, c = n / CHUNK;
    float a = cumA[row];
    float4 yl = *(const float4*)(yloc + i);
    float4 hv = *(const float4*)(Hin + (size_t)(b * NCH + c) * DSD + ch);
    __bf16 o[4] = {(__bf16)(yl.x + a * hv.x), (__bf16)(yl.y + a * hv.y),
                   (__bf16)(yl.z + a * hv.z), (__bf16)(yl.w + a * hv.w)};
    *(uint2*)(ybf + i) = *(const uint2*)o;
}

// ---------------------------------------------------------------------------
extern "C" void kernel_launch(void* const* d_in, const int* in_sizes, int n_in,
                              void* d_out, int out_size, void* d_ws, size_t ws_size,
                              hipStream_t stream) {
    const float* x       = (const float*)d_in[0];
    const float* n1w     = (const float*)d_in[1];
    const float* n1b     = (const float*)d_in[2];
    const float* gate_w  = (const float*)d_in[3];
    const float* gate_b  = (const float*)d_in[4];
    const float* state_w = (const float*)d_in[5];
    const float* state_b = (const float*)d_in[6];
    const float* out_w   = (const float*)d_in[7];
    const float* out_b   = (const float*)d_in[8];
    const float* n2w     = (const float*)d_in[9];
    const float* n2b     = (const float*)d_in[10];
    const float* w1      = (const float*)d_in[11];
    const float* b1      = (const float*)d_in[12];
    const float* w2      = (const float*)d_in[13];
    const float* b2      = (const float*)d_in[14];
    float* out = (float*)d_out;

    // ---- workspace arena (with buffer reuse; ~200MB) ----
    char*  ws  = (char*)d_ws;
    size_t off = 0;
    auto arena = [&](size_t bytes) -> char* {
        char* p = ws + off;
        off += (bytes + 255) & ~(size_t)255;
        return p;
    };
    __bf16* wg_bf = (__bf16*)arena((size_t)DD * 2 * DD * 2);   // gate_w bf16
    __bf16* wsn_bf= (__bf16*)arena((size_t)DD * DSD * 2);      // state_w bf16
    __bf16* wo_bf = (__bf16*)arena((size_t)DSD * DD * 2);      // out_w bf16
    __bf16* w1_bf = (__bf16*)arena((size_t)DD * HH * 2);       // mlp_w1 bf16
    __bf16* w2_bf = (__bf16*)arena((size_t)HH * DD * 2);       // mlp_w2 bf16
    __bf16* xn_bf = (__bf16*)arena((size_t)MM * DD * 2);       // xn; reused as y_bf
    __bf16* xg_bf = (__bf16*)arena((size_t)MM * DD * 2);       // xg; reused as x2n
    __bf16* gg_bf = (__bf16*)arena((size_t)MM * DD * 2);       // gg
    float*  u_f   = (float*)arena((size_t)MM * DSD * 4);       // u -> dBu -> x1
    float*  Bm_f  = (float*)arena((size_t)MM * DSD * 4);       // Bm -> yloc
    __bf16* hm_bf = (__bf16*)arena((size_t)MM * HH * 2);       // mlp hidden
    float*  deltaB= (float*)arena((size_t)MM * 4);
    float*  dAB   = (float*)arena((size_t)MM * 4);
    float*  wmeanB= (float*)arena((size_t)(DD + 1) * 4);
    float*  carryB= (float*)arena((size_t)BB * NCH * DSD * 4);
    float*  HinB  = (float*)arena((size_t)BB * NCH * DSD * 4);
    float*  cumAB = (float*)arena((size_t)BB * NN * 4);
    float*  PB    = (float*)arena((size_t)BB * NCH * 4);
    __bf16* y_bf  = xn_bf;   // reuse (xn dead after gate GEMM + delta)
    __bf16* x2n_bf= xg_bf;   // reuse (xg dead after u GEMM)
    float*  x1_f  = u_f;     // reuse (dBu dead after scan pass 1)
    float*  yloc  = Bm_f;    // reuse (Bm dead after dbu_k)

    // ---- weight conversion to bf16 ----
    auto cvt = [&](const float* s, __bf16* d, int n) {
        cvt_bf16_k<<<(n / 4 + 255) / 256, 256, 0, stream>>>(s, d, n);
    };
    cvt(gate_w,  wg_bf,  DD * 2 * DD);
    cvt(state_w, wsn_bf, DD * DSD);
    cvt(out_w,   wo_bf,  DSD * DD);
    cvt(w1,      w1_bf,  DD * HH);
    cvt(w2,      w2_bf,  HH * DD);

    // ---- xn = LayerNorm1(x) ----
    layernorm_k<<<MM, 256, 0, stream>>>(x, n1w, n1b, xn_bf);

    // ---- wmean (for the collapsed delta matvec) ----
    wmean_k<<<DD + 1, 256, 0, stream>>>(state_w, state_b, wmeanB);

    // ---- gate GEMM: fused SiLU + xn-multiply epilogue -> xg, gg (bf16) ----
    gemm_bf16_k<EPI_GATE><<<dim3(MM / BM, (2 * DD) / BN), 256, 0, stream>>>(
        xn_bf, wg_bf, gate_b, nullptr, xn_bf, xg_bf, gg_bf, MM, 2 * DD, DD);

    // ---- u = xg @ state_w + b ; Bm = gg @ state_w + b ----
    gemm_bf16_k<EPI_F32><<<dim3(MM / BM, DSD / BN), 256, 0, stream>>>(
        xg_bf, wsn_bf, state_b, nullptr, nullptr, u_f, nullptr, MM, DSD, DD);
    gemm_bf16_k<EPI_F32><<<dim3(MM / BM, DSD / BN), 256, 0, stream>>>(
        gg_bf, wsn_bf, state_b, nullptr, nullptr, Bm_f, nullptr, MM, DSD, DD);

    // ---- delta / dA ----
    delta_k<<<MM, 256, 0, stream>>>(xn_bf, wmeanB, deltaB, dAB);

    // ---- dBu = delta * Bm * u (in place over u) ----
    dbu_k<<<(MM * DSD / 4) / 256, 256, 0, stream>>>(u_f, Bm_f, deltaB);

    // ---- chunked parallel scan ----
    cuma_k<<<1, 128, 0, stream>>>(dAB, cumAB, PB);
    scan1_k<<<dim3(NCH, BB), 256, 0, stream>>>(u_f, dAB, yloc, carryB);
    scan2_k<<<BB, 256, 0, stream>>>(carryB, PB, HinB);
    scan3_k<<<(MM * DSD / 4) / 256, 256, 0, stream>>>(yloc, cumAB, HinB, y_bf);

    // ---- x1 = x + y @ out_w + out_b ----
    gemm_bf16_k<EPI_F32><<<dim3(MM / BM, DD / BN), 256, 0, stream>>>(
        y_bf, wo_bf, out_b, x, nullptr, x1_f, nullptr, MM, DD, DSD);

    // ---- x2n = LayerNorm2(x1) ----
    layernorm_k<<<MM, 256, 0, stream>>>(x1_f, n2w, n2b, x2n_bf);

    // ---- hidden = gelu(x2n @ w1 + b1) (bf16) ----
    gemm_bf16_k<EPI_GELU_BF16><<<dim3(MM / BM, HH / BN), 256, 0, stream>>>(
        x2n_bf, w1_bf, b1, nullptr, nullptr, hm_bf, nullptr, MM, HH, DD);

    // ---- out = x1 + hidden @ w2 + b2 ----
    gemm_bf16_k<EPI_F32><<<dim3(MM / BM, DD / BN), 256, 0, stream>>>(
        hm_bf, w2_bf, b2, x1_f, nullptr, out, nullptr, MM, DD, HH);
}
